// MRMLNB_51256139711025
// MI455X (gfx1250) — matrix operand order repaired and checked by reading, not compiled, not run on hardware
//
#include <hip/hip_runtime.h>
#include <hip/hip_bf16.h>
#include <math.h>

typedef __attribute__((ext_vector_type(16))) __bf16 v16bf;
typedef __attribute__((ext_vector_type(8)))  __bf16 v8bf;
typedef __attribute__((ext_vector_type(8)))  float  v8f;

// ---------------------------------------------------------------------------
// f32 -> bf16 elementwise convert
// ---------------------------------------------------------------------------
__global__ void k_f32_to_bf16(const float* __restrict__ src, __bf16* __restrict__ dst, int n) {
    int i = blockIdx.x * blockDim.x + threadIdx.x;
    if (i < n) dst[i] = (__bf16)src[i];
}

// ---------------------------------------------------------------------------
// pe_table[T,8] = pre_table[T,5] @ pe_w[8,5]^T + pe_b  (stored bf16, L2-resident)
// ---------------------------------------------------------------------------
__global__ void k_pe_table(const float* __restrict__ pre, const float* __restrict__ pw,
                           const float* __restrict__ pb, __bf16* __restrict__ out, int T) {
    int t = blockIdx.x * blockDim.x + threadIdx.x;
    if (t >= T) return;
    float p0 = pre[t*5+0], p1 = pre[t*5+1], p2 = pre[t*5+2], p3 = pre[t*5+3], p4 = pre[t*5+4];
#pragma unroll
    for (int j = 0; j < 8; ++j) {
        const float* w = pw + j * 5;
        float s = pb[j] + p0*w[0] + p1*w[1] + p2*w[2] + p3*w[3] + p4*w[4];
        out[(size_t)t*8 + j] = (__bf16)s;
    }
}

// ---------------------------------------------------------------------------
// LayerNorm over last dim (256), one wave32 per row, output bf16
// ---------------------------------------------------------------------------
__global__ void k_layernorm_bf16(const float* __restrict__ src, const float* __restrict__ g,
                                 const float* __restrict__ bta, __bf16* __restrict__ dst, int M) {
    int wid  = threadIdx.x >> 5;
    int lane = threadIdx.x & 31;
    int row  = blockIdx.x * (blockDim.x >> 5) + wid;
    if (row >= M) return;
    const float* p = src + (size_t)row * 256;
    float v[8], s = 0.f, s2 = 0.f;
#pragma unroll
    for (int i = 0; i < 8; ++i) { v[i] = p[i*32 + lane]; s += v[i]; s2 += v[i]*v[i]; }
#pragma unroll
    for (int o = 16; o > 0; o >>= 1) { s += __shfl_xor(s, o, 32); s2 += __shfl_xor(s2, o, 32); }
    float mean = s * (1.f/256.f);
    float var  = s2 * (1.f/256.f) - mean * mean;
    float rstd = rsqrtf(var + 1e-5f);
#pragma unroll
    for (int i = 0; i < 8; ++i) {
        int idx = i*32 + lane;
        dst[(size_t)row*256 + idx] = (__bf16)((v[i] - mean) * rstd * g[idx] + bta[idx]);
    }
}

// ---------------------------------------------------------------------------
// WMMA bf16 GEMM: out[M,N] = A[M,K] @ W[N,K]^T, fused epilogue.
// 8 waves per 256-thread block; each wave owns a 16(M) x 64(N) tile, K-step 32.
// Software-pipelined: next k-step's fragments are loaded while current step's
// 4 WMMAs execute, so the pre-WMMA wait tolerates outstanding prefetch loads.
// Fragment layouts per CDNA5 ISA 7.12.2 (wave32).
// ---------------------------------------------------------------------------
__device__ __forceinline__ v16bf load_a_frag(const __bf16* __restrict__ A, int K,
                                             int mbase, int k0, int lane) {
    int r    = lane & 15;
    int koff = (lane >> 4) << 3;                 // lanes 0-15: K{0..7,16..23}; 16-31: K{8..15,24..31}
    const __bf16* p = A + (size_t)(mbase + r) * K + k0 + koff;
    union { struct { v8bf lo, hi; } h; v16bf v; } u;
    u.h.lo = *reinterpret_cast<const v8bf*>(p);
    u.h.hi = *reinterpret_cast<const v8bf*>(p + 16);
    return u.v;
}

__device__ __forceinline__ v16bf load_b_frag(const __bf16* __restrict__ W, int K,
                                             int nbase, int k0, int lane) {
    int n    = lane & 15;
    int koff = (lane >> 4) << 3;                 // lanes 0-15: K{0..7,16..23}; 16-31: K{8..15,24..31}
    const __bf16* p = W + (size_t)(nbase + n) * K + k0 + koff;
    union { struct { v8bf lo, hi; } h; v16bf v; } u;
    u.h.lo = *reinterpret_cast<const v8bf*>(p);
    u.h.hi = *reinterpret_cast<const v8bf*>(p + 16);
    return u.v;
}

__device__ __forceinline__ float gelu_exact(float x) {
    return 0.5f * x * (1.0f + erff(x * 0.70710678118654752f));
}

__global__ void k_gemm_bf16_wmma(const __bf16* __restrict__ A, const __bf16* __restrict__ W,
                                 const float* __restrict__ bias, const float* __restrict__ residual,
                                 float* __restrict__ outF, __bf16* __restrict__ outB,
                                 int N, int K, float alpha, int do_gelu) {
    int wid   = threadIdx.x >> 5;
    int lane  = threadIdx.x & 31;
    int mbase = blockIdx.x * 128 + wid * 16;
    int nbase = blockIdx.y * 64;

    v8f acc[4] = {};                              // 4 x (16x16) tiles along N

    // --- prologue: fragments for k0 = 0
    v16bf a_cur = load_a_frag(A, K, mbase, 0, lane);
    v16bf b_cur[4];
#pragma unroll
    for (int nt = 0; nt < 4; ++nt) b_cur[nt] = load_b_frag(W, K, nbase + nt * 16, 0, lane);

    // --- pipelined main loop: prefetch k0, compute k0-32
    for (int k0 = 32; k0 < K; k0 += 32) {
        v16bf a_nxt = load_a_frag(A, K, mbase, k0, lane);
        v16bf b_nxt[4];
#pragma unroll
        for (int nt = 0; nt < 4; ++nt) b_nxt[nt] = load_b_frag(W, K, nbase + nt * 16, k0, lane);

#pragma unroll
        for (int nt = 0; nt < 4; ++nt)
            acc[nt] = __builtin_amdgcn_wmma_f32_16x16x32_bf16(
                false, a_cur, false, b_cur[nt], (short)0, acc[nt], false, false);

        a_cur = a_nxt;
#pragma unroll
        for (int nt = 0; nt < 4; ++nt) b_cur[nt] = b_nxt[nt];
    }

    // --- epilogue step: last k-slice
#pragma unroll
    for (int nt = 0; nt < 4; ++nt)
        acc[nt] = __builtin_amdgcn_wmma_f32_16x16x32_bf16(
            false, a_cur, false, b_cur[nt], (short)0, acc[nt], false, false);

    int ln = lane & 15;
    int hi = lane >> 4;                           // D: VGPR j -> row j (lo half) / 8+j (hi half)
#pragma unroll
    for (int nt = 0; nt < 4; ++nt) {
#pragma unroll
        for (int j = 0; j < 8; ++j) {
            int row = mbase + hi * 8 + j;
            int col = nbase + nt * 16 + ln;
            float v = acc[nt][j] * alpha + bias[col];
            if (residual) v += residual[(size_t)row * N + col];
            if (do_gelu)  v = gelu_exact(v);
            if (outB) outB[(size_t)row * N + col] = (__bf16)v;
            else      outF[(size_t)row * N + col] = v;
        }
    }
}

// ---------------------------------------------------------------------------
// Neighborhood attention: one wave32 per (token, head). lane = channel (c_=32).
// Scores held register-distributed (lane==m), softmax over 48 neighbors + blank.
// ---------------------------------------------------------------------------
__global__ void k_attn(const float* __restrict__ q, const float* __restrict__ kv,
                       const int* __restrict__ member_idx, const int* __restrict__ pe_idx,
                       const __bf16* __restrict__ pe_table, const float* __restrict__ mask,
                       const float* __restrict__ blank_k, const float* __restrict__ blank_v,
                       __bf16* __restrict__ out) {
    int tok  = blockIdx.x;
    int hh   = threadIdx.x >> 5;                  // 8 waves per block = 8 heads
    int lane = threadIdx.x & 31;

    float qv = q[(size_t)tok * 256 + hh * 32 + lane];
    const int*   mi = member_idx + (size_t)tok * 48;
    const int*   pi = pe_idx     + (size_t)tok * 48;
    const float* mk = mask       + (size_t)tok * 48;

    float s0 = -1e30f, s1 = -1e30f;               // s0: m=lane, s1: m=32+lane (lanes 0-15), lane16 = blank
    for (int m = 0; m < 48; ++m) {
        int mem = mi[m];
        float d = qv * kv[(size_t)mem * 512 + hh * 64 + lane];
#pragma unroll
        for (int o = 16; o > 0; o >>= 1) d += __shfl_xor(d, o, 32);
        float sc = d + (float)pe_table[(size_t)pi[m] * 8 + hh] + (1.f - mk[m]) * (-100.f);
        if (m < 32) { if (lane == m)      s0 = sc; }
        else        { if (lane == m - 32) s1 = sc; }
    }
    {   // blank score (no pos, no mask)
        float d = qv * blank_k[hh * 32 + lane];
#pragma unroll
        for (int o = 16; o > 0; o >>= 1) d += __shfl_xor(d, o, 32);
        if (lane == 16) s1 = d;
    }

    float mx = fmaxf(s0, s1);
#pragma unroll
    for (int o = 16; o > 0; o >>= 1) mx = fmaxf(mx, __shfl_xor(mx, o, 32));
    float e0 = __expf(s0 - mx), e1 = __expf(s1 - mx);
    float sm = e0 + e1;
#pragma unroll
    for (int o = 16; o > 0; o >>= 1) sm += __shfl_xor(sm, o, 32);
    float inv = 1.f / sm;

    float acc = __shfl(e1, 16, 32) * inv * blank_v[hh * 32 + lane];
    for (int m = 0; m < 48; ++m) {
        float wm = ((m < 32) ? __shfl(e0, m, 32) : __shfl(e1, m - 32, 32)) * inv;
        int mem = mi[m];
        acc += wm * kv[(size_t)mem * 512 + hh * 64 + 32 + lane];
    }
    out[(size_t)tok * 256 + hh * 32 + lane] = (__bf16)acc;
}

// ---------------------------------------------------------------------------
// Orchestration
// ---------------------------------------------------------------------------
extern "C" void kernel_launch(void* const* d_in, const int* in_sizes, int n_in,
                              void* d_out, int out_size, void* d_ws, size_t ws_size,
                              hipStream_t stream) {
    const int c = 256, hid = 512;
    const int Ntok = in_sizes[0] / c;             // b*n = 8192
    const int T    = in_sizes[2] / 5;

    const float* feat       = (const float*)d_in[0];
    const float* cmask      = (const float*)d_in[1];
    const float* pre_table  = (const float*)d_in[2];
    const float* norm1_w    = (const float*)d_in[3];
    const float* norm1_b    = (const float*)d_in[4];
    const float* q_w        = (const float*)d_in[5];
    const float* q_b        = (const float*)d_in[6];
    const float* kv_w       = (const float*)d_in[7];
    const float* kv_b       = (const float*)d_in[8];
    const float* blank_k    = (const float*)d_in[9];
    const float* blank_v    = (const float*)d_in[10];
    const float* pe_w       = (const float*)d_in[11];
    const float* pe_b       = (const float*)d_in[12];
    const float* proj_w     = (const float*)d_in[13];
    const float* proj_b     = (const float*)d_in[14];
    const float* norm2_w    = (const float*)d_in[15];
    const float* norm2_b    = (const float*)d_in[16];
    const float* fc1_w      = (const float*)d_in[17];
    const float* fc1_b      = (const float*)d_in[18];
    const float* fc2_w      = (const float*)d_in[19];
    const float* fc2_b      = (const float*)d_in[20];
    const int*   member_idx = (const int*)d_in[21];
    const int*   pe_idx     = (const int*)d_in[22];

    // workspace carve (256B aligned)
    char* wp = (char*)d_ws;
    auto carve = [&](size_t bytes) -> void* {
        void* p = (void*)wp;
        wp += (bytes + 255) & ~(size_t)255;
        return p;
    };
    __bf16* x_bf    = (__bf16*)carve((size_t)Ntok * c * 2);
    __bf16* qw_bf   = (__bf16*)carve((size_t)c * c * 2);
    __bf16* kvw_bf  = (__bf16*)carve((size_t)2 * c * c * 2);
    __bf16* pjw_bf  = (__bf16*)carve((size_t)c * c * 2);
    __bf16* f1w_bf  = (__bf16*)carve((size_t)hid * c * 2);
    __bf16* f2w_bf  = (__bf16*)carve((size_t)c * hid * 2);
    float*  qbuf    = (float*) carve((size_t)Ntok * c * 4);
    float*  kvbuf   = (float*) carve((size_t)Ntok * 2 * c * 4);
    __bf16* petab   = (__bf16*)carve((size_t)T * 8 * 2);
    __bf16* attnout = (__bf16*)carve((size_t)Ntok * c * 2);
    float*  feat2   = (float*) carve((size_t)Ntok * c * 4);
    __bf16* y_bf    = (__bf16*)carve((size_t)Ntok * c * 2);
    __bf16* h1_bf   = (__bf16*)carve((size_t)Ntok * hid * 2);
    (void)ws_size; (void)n_in; (void)out_size;

    // 1) weight conversions + pe_table
    auto cvt = [&](const float* s, __bf16* d, int n) {
        k_f32_to_bf16<<<(n + 255) / 256, 256, 0, stream>>>(s, d, n);
    };
    cvt(q_w,    qw_bf,  c * c);
    cvt(kv_w,   kvw_bf, 2 * c * c);
    cvt(proj_w, pjw_bf, c * c);
    cvt(fc1_w,  f1w_bf, hid * c);
    cvt(fc2_w,  f2w_bf, c * hid);
    k_pe_table<<<(T + 255) / 256, 256, 0, stream>>>(pre_table, pe_w, pe_b, petab, T);

    // 2) LN1
    k_layernorm_bf16<<<(Ntok + 7) / 8, 256, 0, stream>>>(feat, norm1_w, norm1_b, x_bf, Ntok);

    // 3) q = LN(x) @ q_w^T * c_^-0.5 + q_b ; kv = LN(x) @ kv_w^T + kv_b
    const float alpha_q = 0.17677669529663687f;   // 1/sqrt(32)
    {
        dim3 g(Ntok / 128, c / 64);
        k_gemm_bf16_wmma<<<g, 256, 0, stream>>>(x_bf, qw_bf, q_b, nullptr, qbuf, nullptr,
                                                c, c, alpha_q, 0);
    }
    {
        dim3 g(Ntok / 128, (2 * c) / 64);
        k_gemm_bf16_wmma<<<g, 256, 0, stream>>>(x_bf, kvw_bf, kv_b, nullptr, kvbuf, nullptr,
                                                2 * c, c, 1.0f, 0);
    }

    // 4) attention (one block per token, 8 waves = 8 heads)
    k_attn<<<Ntok, 256, 0, stream>>>(qbuf, kvbuf, member_idx, pe_idx, petab, cmask,
                                     blank_k, blank_v, attnout);

    // 5) feat2 = feat + attnout @ proj_w^T + proj_b
    {
        dim3 g(Ntok / 128, c / 64);
        k_gemm_bf16_wmma<<<g, 256, 0, stream>>>(attnout, pjw_bf, proj_b, feat, feat2, nullptr,
                                                c, c, 1.0f, 0);
    }

    // 6) LN2
    k_layernorm_bf16<<<(Ntok + 7) / 8, 256, 0, stream>>>(feat2, norm2_w, norm2_b, y_bf, Ntok);

    // 7) h1 = gelu(y @ fc1_w^T + fc1_b)  (bf16)
    {
        dim3 g(Ntok / 128, hid / 64);
        k_gemm_bf16_wmma<<<g, 256, 0, stream>>>(y_bf, f1w_bf, fc1_b, nullptr, nullptr, h1_bf,
                                                hid, c, 1.0f, 1);
    }

    // 8) out = feat2 + h1 @ fc2_w^T + fc2_b
    {
        dim3 g(Ntok / 128, c / 64);
        k_gemm_bf16_wmma<<<g, 256, 0, stream>>>(h1_bf, f2w_bf, fc2_b, feat2, (float*)d_out, nullptr,
                                                c, hid, 1.0f, 0);
    }
}
